// TransformerDecoder_47802986004690
// MI455X (gfx1250) — matrix operand order repaired and checked by reading, not compiled, not run on hardware
//
#include <hip/hip_runtime.h>
#include <math.h>

// ---------------- types ----------------
typedef unsigned short u16;
typedef __attribute__((ext_vector_type(16))) __bf16 v16bf;
typedef __attribute__((ext_vector_type(8)))  float  v8f;
typedef __attribute__((ext_vector_type(4)))  int    v4i_;

#define WMMA_BF16(a, b, c) \
  __builtin_amdgcn_wmma_f32_16x16x32_bf16(false, (a), false, (b), (short)0, (c), false, false)

// ---------------- CDNA5 async global->LDS (guarded) ----------------
#if defined(__has_builtin)
#if __has_builtin(__builtin_amdgcn_global_load_async_to_lds_b128) && \
    __has_builtin(__builtin_amdgcn_s_wait_asynccnt)
#define HAS_ASYNC 1
#endif
#endif
#ifndef HAS_ASYNC
#define HAS_ASYNC 0
#endif

#if HAS_ASYNC
// toolchain declares params as generic int4* (per diagnostic); order: (global src, lds dst, imm off, imm cpol)
#define GLD_ASYNC_B128(gp, lp)                                             \
  __builtin_amdgcn_global_load_async_to_lds_b128(                          \
      (v4i_*)(void*)(gp), (v4i_*)(void*)(lp), 0, 0)
#define WAIT_ASYNC(n) __builtin_amdgcn_s_wait_asynccnt(n)
#define COPY16(gp, lp) GLD_ASYNC_B128(gp, lp)
#else
#define WAIT_ASYNC(n)
#define COPY16(gp, lp) (*(uint4*)(lp) = *(const uint4*)(gp))
#endif

union FragU { v16bf v; uint4 u[2]; };

// f32 -> bf16 (round-to-nearest-even) as raw u16
__device__ __forceinline__ u16 f2bf(float x) {
  unsigned int u = __float_as_uint(x);
  unsigned int r = (u + 0x7FFFu + ((u >> 16) & 1u)) >> 16;
  return (u16)r;
}

// A fragment (16x32 bf16):
//  lanes 0-15 : row=lane,    K = {k0..k0+7} U {k0+16..k0+23}
//  lanes 16-31: row=lane-16, K = {k0+8..k0+15} U {k0+24..k0+31}
__device__ __forceinline__ v16bf load_frag_a(const u16* base, int stride, int k0, int lane) {
  int hf = (lane >> 4) & 1, r = lane & 15;
  const u16* p = base + r * stride + k0 + hf * 8;
  FragU f;
  f.u[0] = *(const uint4*)(p);
  f.u[1] = *(const uint4*)(p + 16);
  return f.v;
}

// B fragment (32x16 bf16) from transposed tile [n][k]:
//  lanes 0-15 : col=lane,    K = {k0..k0+7} U {k0+16..k0+23}
//  lanes 16-31: col=lane-16, K = {k0+8..k0+15} U {k0+24..k0+31}
__device__ __forceinline__ v16bf load_frag_b(const u16* base, int stride, int k0, int lane) {
  int hf = (lane >> 4) & 1, r = lane & 15;
  const u16* p = base + r * stride + k0 + hf * 8;
  FragU f;
  f.u[0] = *(const uint4*)(p);
  f.u[1] = *(const uint4*)(p + 16);
  return f.v;
}

// ---------------- f32 [K,M] -> bf16 transposed [M,K] ----------------
__global__ void cvt_w_t(const float* __restrict__ in, u16* __restrict__ out,
                        int Kd, int Md) {
  int i = blockIdx.x * blockDim.x + threadIdx.x;
  if (i < Kd * Md) {
    int k = i / Md, m = i - k * Md;
    out[(size_t)m * Kd + k] = f2bf(in[i]);
  }
}

// ---------------- RMSNorm: one wave32 per row ----------------
__launch_bounds__(256)
__global__ void rmsnorm_bf16(const float* __restrict__ x, const float* __restrict__ w,
                             u16* __restrict__ out, int D) {
  int wave = threadIdx.x >> 5, lane = threadIdx.x & 31;
  int row  = blockIdx.x * 8 + wave;
  const float* xr = x + (size_t)row * D;
  float ss = 0.f;
  for (int j = lane; j < D; j += 32) { float v = xr[j]; ss += v * v; }
#pragma unroll
  for (int m = 16; m >= 1; m >>= 1) ss += __shfl_xor(ss, m, 32);
  float inv = 1.f / (sqrtf(ss / (float)D) + 1e-6f);   // eps OUTSIDE sqrt, per reference
  u16* orow = out + (size_t)row * D;
  for (int j = lane; j < D; j += 32) orow[j] = f2bf(w[j] * xr[j] * inv);
}

// ---------------- bf16 WMMA GEMM: A[N,K] x Bt[M,K]^T ----------------
// MODE 0: out_bf16 = acc + bias                       (Q,K proj)
// MODE 1: out_f32  = acc + bias                       (FFN w0 -> g0)
// MODE 2: out_bf16 = aux * silu(acc + bias)           (FFN w1 gate, aux = g0)
// MODE 3: out_f32 += acc + bias  (in-place residual)  (O proj, FFN w2)
// MODE 4: out_bf16 = acc + bias, written V-transposed [b][h][d][s]
#define BM 128
#define BN 128
#define BK 32
#define LSTR 40   // LDS row stride in u16 (80 bytes, 16B aligned)

template <int MODE>
__launch_bounds__(256)
__global__ void gemm_bf16(const u16* __restrict__ A, const u16* __restrict__ Bt,
                          const float* __restrict__ bias, const float* __restrict__ aux,
                          float* outF, u16* outB, int N, int K, int M) {
  __shared__ __align__(16) u16 As[2][BM * LSTR];
  __shared__ __align__(16) u16 Bs[2][BN * LSTR];
  int t = threadIdx.x, lane = t & 31, wave = t >> 5;
  int n0 = blockIdx.y * BM, m0 = blockIdx.x * BN;
  int rowBase = (wave >> 1) * 32;   // 4 row groups of 32
  int colBase = (wave & 1) * 64;    // 2 col groups of 64
  v8f acc[2][4] = {};

  // per-thread tile mapping: 512 16B chunks per tile, 2 per thread
  int rA0 = t >> 2,         cA0 = (t & 3) * 8;
  int rA1 = (t + 256) >> 2, cA1 = cA0;   // (t+256)&3 == t&3

  auto issue = [&](int buf, int k0) {
    COPY16(A  + (size_t)(n0 + rA0) * K + k0 + cA0, &As[buf][rA0 * LSTR + cA0]);
    COPY16(Bt + (size_t)(m0 + rA0) * K + k0 + cA0, &Bs[buf][rA0 * LSTR + cA0]);
    COPY16(A  + (size_t)(n0 + rA1) * K + k0 + cA1, &As[buf][rA1 * LSTR + cA1]);
    COPY16(Bt + (size_t)(m0 + rA1) * K + k0 + cA1, &Bs[buf][rA1 * LSTR + cA1]);
  };

  issue(0, 0);
  int nk = K / BK;
  for (int ki = 0; ki < nk; ++ki) {
    int cur = ki & 1;
    if (ki + 1 < nk) {
      issue(1 - cur, (ki + 1) * BK);   // prefetch next tile into other buffer
      WAIT_ASYNC(4);                   // allow the 4 just-issued to stay in flight
    } else {
      WAIT_ASYNC(0);
    }
    __syncthreads();
    v16bf a0 = load_frag_a(&As[cur][(size_t)rowBase * LSTR], LSTR, 0, lane);
    v16bf a1 = load_frag_a(&As[cur][(size_t)(rowBase + 16) * LSTR], LSTR, 0, lane);
#pragma unroll
    for (int cj = 0; cj < 4; ++cj) {
      v16bf b = load_frag_b(&Bs[cur][(size_t)(colBase + cj * 16) * LSTR], LSTR, 0, lane);
      acc[0][cj] = WMMA_BF16(a0, b, acc[0][cj]);
      acc[1][cj] = WMMA_BF16(a1, b, acc[1][cj]);
    }
    __syncthreads();
  }

  int hf = lane >> 4, r = lane & 15;
#pragma unroll
  for (int ri = 0; ri < 2; ++ri)
#pragma unroll
    for (int cj = 0; cj < 4; ++cj) {
      int col = m0 + colBase + cj * 16 + r;
      float bv = bias[col];
#pragma unroll
      for (int i = 0; i < 8; ++i) {
        int row = n0 + rowBase + ri * 16 + i + 8 * hf;  // C layout: VGPR i -> M=i / i+8
        float v = acc[ri][cj][i] + bv;
        if (MODE == 0) {
          outB[(size_t)row * M + col] = f2bf(v);
        } else if (MODE == 1) {
          outF[(size_t)row * M + col] = v;
        } else if (MODE == 2) {
          size_t o = (size_t)row * M + col;
          float s = v / (1.f + __expf(-v));
          outB[o] = f2bf(aux[o] * s);
        } else if (MODE == 3) {
          size_t o = (size_t)row * M + col;
          outF[o] += v;
        } else {  // MODE 4: V transposed [b][h][d][s], S=2048, H=8, hd=64
          int bb = row >> 11, s = row & 2047, hh = col >> 6, d = col & 63;
          outB[((((size_t)bb * 8 + hh) * 64 + d) << 11) + s] = f2bf(v);
        }
      }
    }
}

// ---------------- Flash attention (bf16 WMMA, online softmax) ----------------
// 4 waves, 64 q-rows per block, 32-key tiles, hd=64, causal.
// V comes pre-transposed: vT[b][h][d][s].
#define HD 64
__launch_bounds__(128)
__global__ void flash_attn(const u16* __restrict__ Q, const u16* __restrict__ Kb,
                           const u16* __restrict__ Vt_g, u16* __restrict__ O,
                           int S, int H) {
  __shared__ __align__(16) u16 Qs[64 * 72];       // [qrow][hd]
  __shared__ __align__(16) u16 Ks[32 * 72];       // [key][hd]
  __shared__ __align__(16) u16 Vt[64 * 40];       // [hd][key]
  __shared__ __align__(16) u16 Ps[4][16 * 40];    // per-wave P tile [qrow][key]

  int t = threadIdx.x, lane = t & 31, wave = t >> 5;
  int hf = lane >> 4, r = lane & 15;
  int qb0 = blockIdx.x * 64;
  int bh  = blockIdx.y, b = bh / H, h = bh % H;
  int Dm  = H * HD;                                // 512
  size_t nbase = (size_t)b * S + qb0;
  size_t vbase = (size_t)bh * HD * S;              // vT[b][h] block

  // load Q block (64x64): 512 chunks, 4 per thread
#pragma unroll
  for (int j = 0; j < 4; ++j) {
    int c = t + 128 * j;
    int row = c >> 3, cg = (c & 7) * 8;
    COPY16(Q + (nbase + row) * Dm + h * HD + cg, &Qs[row * 72 + cg]);
  }

  v8f o0 = {}, o1 = {}, o2 = {}, o3 = {};
  float rmax[8], rsum[8];
#pragma unroll
  for (int i = 0; i < 8; ++i) { rmax[i] = -1e30f; rsum[i] = 0.f; }

  int kend = qb0 + 64;
  for (int kb = 0; kb < kend; kb += 32) {
    __syncthreads();   // previous iteration done reading Ks/Vt
    // K tile (32x64) and V^T tile (64x32), 2+2 contiguous 16B chunks per thread
#pragma unroll
    for (int j = 0; j < 2; ++j) {
      int c = t + 128 * j;
      int krow = c >> 3, kcg = (c & 7) * 8;
      COPY16(Kb + ((size_t)b * S + kb + krow) * Dm + h * HD + kcg, &Ks[krow * 72 + kcg]);
      int d = c >> 2, vcg = (c & 3) * 8;
      COPY16(Vt_g + vbase + (size_t)d * S + kb + vcg, &Vt[d * 40 + vcg]);
    }
    WAIT_ASYNC(0);
    __syncthreads();

    // scores S = Q * K^T (16x32 per wave, two 16x16 tiles)
    v8f s0 = {}, s1 = {};
#pragma unroll
    for (int kk = 0; kk < HD; kk += 32) {
      v16bf qf  = load_frag_a(Qs + wave * 16 * 72, 72, kk, lane);
      v16bf kf0 = load_frag_b(Ks, 72, kk, lane);
      v16bf kf1 = load_frag_b(Ks + 16 * 72, 72, kk, lane);
      s0 = WMMA_BF16(qf, kf0, s0);
      s1 = WMMA_BF16(qf, kf1, s1);
    }

    // online softmax (C layout: reg i -> qrow i / i+8, column = lane&15)
    int qrowb = qb0 + wave * 16 + 8 * hf;
    int key0 = kb + r, key1 = kb + 16 + r;
    float p0s[8], p1s[8];
#pragma unroll
    for (int i = 0; i < 8; ++i) {
      int qrow = qrowb + i;
      float v0 = s0[i] * 0.125f;           // 1/sqrt(64)
      float v1 = s1[i] * 0.125f;
      if (key0 > qrow) v0 = -1e30f;
      if (key1 > qrow) v1 = -1e30f;
      float mx = fmaxf(v0, v1);
#pragma unroll
      for (int m = 8; m >= 1; m >>= 1) mx = fmaxf(mx, __shfl_xor(mx, m, 32));
      float mnew = fmaxf(rmax[i], mx);
      float corr = __expf(rmax[i] - mnew);
      float p0 = __expf(v0 - mnew), p1 = __expf(v1 - mnew);
      float ps = p0 + p1;
#pragma unroll
      for (int m = 8; m >= 1; m >>= 1) ps += __shfl_xor(ps, m, 32);
      rsum[i] = rsum[i] * corr + ps;
      rmax[i] = mnew;
      o0[i] *= corr; o1[i] *= corr; o2[i] *= corr; o3[i] *= corr;
      p0s[i] = p0; p1s[i] = p1;
    }

    // C-layout -> A-layout for P via per-wave LDS round trip
    u16* pw = Ps[wave];
#pragma unroll
    for (int i = 0; i < 8; ++i) {
      pw[(i + 8 * hf) * 40 + r]      = f2bf(p0s[i]);
      pw[(i + 8 * hf) * 40 + 16 + r] = f2bf(p1s[i]);
    }
    __builtin_amdgcn_wave_barrier();
    asm volatile("s_wait_dscnt 0" ::: "memory");

    v16bf pf  = load_frag_a(pw, 40, 0, lane);
    v16bf vf0 = load_frag_b(Vt,           40, 0, lane);
    v16bf vf1 = load_frag_b(Vt + 16 * 40, 40, 0, lane);
    v16bf vf2 = load_frag_b(Vt + 32 * 40, 40, 0, lane);
    v16bf vf3 = load_frag_b(Vt + 48 * 40, 40, 0, lane);
    o0 = WMMA_BF16(pf, vf0, o0);
    o1 = WMMA_BF16(pf, vf1, o1);
    o2 = WMMA_BF16(pf, vf2, o2);
    o3 = WMMA_BF16(pf, vf3, o3);
  }

  // normalize and write attention output (bf16, head-interleaved [n][h*64+d])
#pragma unroll
  for (int i = 0; i < 8; ++i) {
    float inv = 1.f / rsum[i];
    size_t row = nbase + wave * 16 + i + 8 * hf;
    u16* op = O + row * Dm + h * HD;
    op[0 * 16 + r] = f2bf(o0[i] * inv);
    op[1 * 16 + r] = f2bf(o1[i] * inv);
    op[2 * 16 + r] = f2bf(o2[i] * inv);
    op[3 * 16 + r] = f2bf(o3[i] * inv);
  }
}

// ---------------- host orchestration ----------------
extern "C" void kernel_launch(void* const* d_in, const int* in_sizes, int n_in,
                              void* d_out, int out_size, void* d_ws, size_t ws_size,
                              hipStream_t stream) {
  const int L = 4, D = 512, S = 2048, B = 2, H = 8, HID = 2048;
  const int N = B * S;

  const float* x   = (const float*)d_in[0];
  const float* qw  = (const float*)d_in[1];
  const float* qb  = (const float*)d_in[2];
  const float* kw  = (const float*)d_in[3];
  const float* kbv = (const float*)d_in[4];
  const float* vw  = (const float*)d_in[5];
  const float* vb  = (const float*)d_in[6];
  const float* ow  = (const float*)d_in[7];
  const float* ob  = (const float*)d_in[8];
  const float* w0  = (const float*)d_in[9];
  const float* b0  = (const float*)d_in[10];
  const float* w1  = (const float*)d_in[11];
  const float* b1  = (const float*)d_in[12];
  const float* w2  = (const float*)d_in[13];
  const float* b2  = (const float*)d_in[14];
  const float* anw = (const float*)d_in[15];
  const float* fnw = (const float*)d_in[16];

  char* ws = (char*)d_ws;
  size_t off = 0;
  auto alloc = [&](size_t bytes) -> void* {
    off = (off + 255) & ~(size_t)255;
    void* p = ws + off;
    off += bytes;
    return p;
  };
  u16* uqw = (u16*)alloc((size_t)D * D * 2);     // transposed bf16 weights
  u16* ukw = (u16*)alloc((size_t)D * D * 2);
  u16* uvw = (u16*)alloc((size_t)D * D * 2);
  u16* uow = (u16*)alloc((size_t)D * D * 2);
  u16* uw0 = (u16*)alloc((size_t)D * HID * 2);
  u16* uw1 = (u16*)alloc((size_t)D * HID * 2);
  u16* uw2 = (u16*)alloc((size_t)HID * D * 2);
  u16* xn  = (u16*)alloc((size_t)N * D * 2);
  u16* qbf = (u16*)alloc((size_t)N * D * 2);
  u16* kbf = (u16*)alloc((size_t)N * D * 2);
  u16* vT  = (u16*)alloc((size_t)N * D * 2);     // [b][h][d][s]
  u16* att = (u16*)alloc((size_t)N * D * 2);
  float* g0 = (float*)alloc((size_t)N * HID * 4);
  u16* ffa = (u16*)alloc((size_t)N * HID * 2);

  float* hbuf = (float*)d_out;   // hidden state lives in d_out (fp32), updated in place
  (void)hipMemcpyAsync(hbuf, x, (size_t)N * D * sizeof(float), hipMemcpyDeviceToDevice, stream);

  dim3 gD(D / BN, N / BM);
  dim3 gH(HID / BN, N / BM);
  dim3 gA(S / 64, B * H);

  for (int l = 0; l < L; ++l) {
    int nDD = D * D, nDH = D * HID;
    // convert + transpose this layer's weights to bf16 [M][K]
    cvt_w_t<<<(nDD + 255) / 256, 256, 0, stream>>>(qw + (size_t)l * nDD, uqw, D, D);
    cvt_w_t<<<(nDD + 255) / 256, 256, 0, stream>>>(kw + (size_t)l * nDD, ukw, D, D);
    cvt_w_t<<<(nDD + 255) / 256, 256, 0, stream>>>(vw + (size_t)l * nDD, uvw, D, D);
    cvt_w_t<<<(nDD + 255) / 256, 256, 0, stream>>>(ow + (size_t)l * nDD, uow, D, D);
    cvt_w_t<<<(nDH + 255) / 256, 256, 0, stream>>>(w0 + (size_t)l * nDH, uw0, D, HID);
    cvt_w_t<<<(nDH + 255) / 256, 256, 0, stream>>>(w1 + (size_t)l * nDH, uw1, D, HID);
    cvt_w_t<<<(nDH + 255) / 256, 256, 0, stream>>>(w2 + (size_t)l * nDH, uw2, HID, D);

    // attention block
    rmsnorm_bf16<<<N / 8, 256, 0, stream>>>(hbuf, anw + (size_t)l * D, xn, D);
    gemm_bf16<0><<<gD, 256, 0, stream>>>(xn, uqw, qb  + (size_t)l * D, nullptr, nullptr, qbf, N, D, D);
    gemm_bf16<0><<<gD, 256, 0, stream>>>(xn, ukw, kbv + (size_t)l * D, nullptr, nullptr, kbf, N, D, D);
    gemm_bf16<4><<<gD, 256, 0, stream>>>(xn, uvw, vb  + (size_t)l * D, nullptr, nullptr, vT,  N, D, D);
    flash_attn<<<gA, 128, 0, stream>>>(qbf, kbf, vT, att, S, H);
    gemm_bf16<3><<<gD, 256, 0, stream>>>(att, uow, ob + (size_t)l * D, nullptr, hbuf, nullptr, N, D, D);

    // FFN block
    rmsnorm_bf16<<<N / 8, 256, 0, stream>>>(hbuf, fnw + (size_t)l * D, xn, D);
    gemm_bf16<1><<<gH, 256, 0, stream>>>(xn, uw0, b0 + (size_t)l * HID, nullptr, g0, nullptr, N, D, HID);
    gemm_bf16<2><<<gH, 256, 0, stream>>>(xn, uw1, b1 + (size_t)l * HID, g0, nullptr, ffa, N, D, HID);
    gemm_bf16<3><<<gD, 256, 0, stream>>>(ffa, uw2, b2 + (size_t)l * D, nullptr, hbuf, nullptr, N, HID, D);
  }
  (void)in_sizes; (void)n_in; (void)out_size; (void)ws_size;
}
